// GraphSAGEPredictor_21500606284208
// MI455X (gfx1250) — matrix operand order repaired and checked by reading, not compiled, not run on hardware
//
#include <hip/hip_runtime.h>
#include <hip/hip_fp16.h>

typedef __attribute__((ext_vector_type(16))) _Float16 v16h;
typedef __attribute__((ext_vector_type(8)))  _Float16 v8h;
typedef __attribute__((ext_vector_type(4)))  _Float16 v4h;
typedef __attribute__((ext_vector_type(8)))  float    v8f;

#define NN      100000
#define NE      1600000
#define NG      1024
#define IN_F    74
#define IN_P    80        // padded input feature width
#define HID     100
#define HID_P   112       // padded hidden width (7 * 16)
#define KTOT1   160       // 80 self + 80 neigh
#define KTOT2   224       // 112 self + 112 neigh
#define NCT     7         // 112 / 16 output col tiles
#define MT      (NN/16)   // 6250 row tiles (exact)

// ---------------- prep kernels ----------------

// pad node features to [NN][80] and convert to f16 (GEMM consumes f16 anyway)
__global__ void pad_x_kernel(const float* __restrict__ X, _Float16* __restrict__ Xp) {
    int t = blockIdx.x * blockDim.x + threadIdx.x;
    if (t >= NN * IN_P) return;
    int m = t / IN_P, k = t - m * IN_P;
    Xp[t] = (_Float16)((k < IN_F) ? X[m * IN_F + k] : 0.f);
}

// Build transposed, zero-padded f16 weight blocks: Wt[n*Ktot + k]
__global__ void prep_w_kernel(const float* __restrict__ Ws1, const float* __restrict__ Wn1,
                              const float* __restrict__ Ws2, const float* __restrict__ Wn2,
                              _Float16* __restrict__ Wt1, _Float16* __restrict__ Wt2) {
    int t = blockIdx.x * blockDim.x + threadIdx.x;
    if (t < HID_P * KTOT1) {
        int n = t / KTOT1, k = t - n * KTOT1;
        float v = 0.f;
        if (n < HID) {
            if (k < IN_F)                          v = Ws1[k * HID + n];
            else if (k >= IN_P && k < IN_P + IN_F) v = Wn1[(k - IN_P) * HID + n];
        }
        Wt1[t] = (_Float16)v;
    }
    if (t < HID_P * KTOT2) {
        int n = t / KTOT2, k = t - n * KTOT2;
        float v = 0.f;
        if (n < HID) {
            if (k < HID)                             v = Ws2[k * HID + n];
            else if (k >= HID_P && k < HID_P + HID)  v = Wn2[(k - HID_P) * HID + n];
        }
        Wt2[t] = (_Float16)v;
    }
}

__global__ void deg_count_kernel(const int* __restrict__ dst, float* __restrict__ deg) {
    int t = blockIdx.x * blockDim.x + threadIdx.x;
    if (t < NE) atomicAdd(&deg[dst[t]], 1.f);
}

__global__ void rdeg_kernel(float* __restrict__ deg) {
    int t = blockIdx.x * blockDim.x + threadIdx.x;
    if (t < NN) deg[t] = 1.f / fmaxf(deg[t], 1.f);
}

// ---------------- edge scatter: segment_sum(feat_f16[src]) -> agg_f32[dst] ----------

__global__ void scatter_kernel(const _Float16* __restrict__ feat, int ldf, int nc4,
                               const int* __restrict__ src, const int* __restrict__ dst,
                               float* __restrict__ agg, int lda, int total) {
    int t = blockIdx.x * blockDim.x + threadIdx.x;
    if (t >= total) return;
    int e = t / nc4;
    int c = (t - e * nc4) * 4;
    int s = src[e], d = dst[e];
    v4h v = *(const v4h*)(feat + (size_t)s * ldf + c);   // 8-byte load
    float* ap = agg + (size_t)d * lda + c;
    atomicAdd(ap + 0, (float)v[0]);
    atomicAdd(ap + 1, (float)v[1]);
    atomicAdd(ap + 2, (float)v[2]);
    atomicAdd(ap + 3, (float)v[3]);
}

// ---------------- fused SAGE WMMA GEMM ----------------
// Out[m,:112] = relu( A1_f16[m,:Kp1]@W + (A2_f32[m,:]*rdeg[m])@W + bias ), pad cols -> 0
// Wt is [112][Ktot] f16 transposed so a B fragment is two runs of 8 contiguous halves (k and k+16) per lane.

__device__ __forceinline__ void fill8h(_Float16* o, const _Float16* p) {
    *(v8h*)o = *(const v8h*)p;                            // 16-byte copy
}

__device__ __forceinline__ void fill8f(_Float16* o, const float* p, float sc) {
    float4 v0 = *(const float4*)p;
    float4 v1 = *(const float4*)(p + 4);
    o[0] = (_Float16)(v0.x * sc); o[1] = (_Float16)(v0.y * sc);
    o[2] = (_Float16)(v0.z * sc); o[3] = (_Float16)(v0.w * sc);
    o[4] = (_Float16)(v1.x * sc); o[5] = (_Float16)(v1.y * sc);
    o[6] = (_Float16)(v1.z * sc); o[7] = (_Float16)(v1.w * sc);
}

template <typename TOUT>
__global__ __launch_bounds__(256) void sage_gemm_kernel(
        const _Float16* __restrict__ A1, int lda1, int Kp1,
        const float* __restrict__ A2, int lda2,
        const float* __restrict__ rdeg,
        const _Float16* __restrict__ Wt, int Ktot,
        const float* __restrict__ bias,
        TOUT* __restrict__ Out, int Nvalid) {
    int wid = blockIdx.x * 8 + (threadIdx.x >> 5);
    if (wid >= MT) return;
    int lane  = threadIdx.x & 31;
    int lan16 = lane & 15;
    int m     = wid * 16 + lan16;                // A row for this lane
    int kb8   = (lane < 16) ? 0 : 8;             // A-fragment K base
    int kbB   = (lane < 16) ? 0 : 8;             // B-fragment K base
    float s   = rdeg[m];
    const _Float16* a1row = A1 + (size_t)m * lda1;
    const float*    a2row = A2 + (size_t)m * lda2;
    const _Float16* wl    = Wt + (size_t)lan16 * Ktot + kbB;  // ct stride = 16*Ktot

    v8f acc[NCT] = {};
    union { v16h v; _Float16 e[16]; } a;

    for (int kk = 0; kk < Ktot; kk += 32) {
        // Wave-uniform selects (Kp1, kk multiples of 16; lane offset < 16)
        if (kk < Kp1)      fill8h(a.e,     a1row + kk + kb8);
        else               fill8f(a.e,     a2row + (kk + kb8 - Kp1), s);
        if (kk + 16 < Kp1) fill8h(a.e + 8, a1row + kk + 16 + kb8);
        else               fill8f(a.e + 8, a2row + (kk + 16 + kb8 - Kp1), s);

        // load all 7 B fragments first so WMMAs don't each wait on loadcnt 0
        v16h bf[NCT];
        #pragma unroll
        for (int ct = 0; ct < NCT; ct++)
            bf[ct] = __builtin_shufflevector(*(const v8h*)(wl + kk + (size_t)ct * 16 * Ktot), *(const v8h*)(wl + kk + 16 + (size_t)ct * 16 * Ktot), 0, 1, 2, 3, 4, 5, 6, 7, 8, 9, 10, 11, 12, 13, 14, 15);
        #pragma unroll
        for (int ct = 0; ct < NCT; ct++)
            acc[ct] = __builtin_amdgcn_wmma_f32_16x16x32_f16(
                false, a.v, false, bf[ct], (short)0, acc[ct], false, false);
    }

    // epilogue: lane covers col n; VGPR i -> row rbase + i
    int rbase = wid * 16 + ((lane < 16) ? 0 : 8);
    #pragma unroll
    for (int ct = 0; ct < NCT; ct++) {
        int n = ct * 16 + lan16;
        float bn = (n < Nvalid) ? bias[n] : 0.f;
        #pragma unroll
        for (int i = 0; i < 8; i++) {
            float v = fmaxf(acc[ct][i] + bn, 0.f);
            if (n >= Nvalid) v = 0.f;
            Out[(size_t)(rbase + i) * HID_P + n] = (TOUT)v;
        }
    }
}

// ---------------- WeightedSumAndMax readout ----------------

__global__ void readout_kernel(const float* __restrict__ h2, const int* __restrict__ gid,
                               const float* __restrict__ watom, const float* __restrict__ batom,
                               float* __restrict__ gf) {
    int node = blockIdx.x * 8 + (threadIdx.x >> 5);
    if (node >= NN) return;
    int lane = threadIdx.x & 31;
    const float* hp = h2 + (size_t)node * HID_P;
    float v[4]; float dot = 0.f;
    #pragma unroll
    for (int j = 0; j < 4; j++) {
        int c = lane + 32 * j;
        float x = (c < HID) ? hp[c] : 0.f;
        v[j] = x;
        if (c < HID) dot += x * watom[c];
    }
    #pragma unroll
    for (int off = 16; off > 0; off >>= 1) dot += __shfl_xor(dot, off, 32);
    float w = 1.f / (1.f + expf(-(dot + batom[0])));
    int g = gid[node];
    float* gs = gf + (size_t)g * (2 * HID);
    #pragma unroll
    for (int j = 0; j < 4; j++) {
        int c = lane + 32 * j;
        if (c < HID) {
            atomicAdd(gs + c, v[j] * w);
            // h2 >= 0 (ReLU): int-compare max == float max; 0-init matches neginf->0
            atomicMax((int*)(gs + HID + c), __float_as_int(v[j]));
        }
    }
}

// ---------------- final MLP: (gf @ Wp1 + bp1) @ Wp2 + bp2 ----------------

__global__ void mlp_kernel(const float* __restrict__ gf,
                           const float* __restrict__ Wp1, const float* __restrict__ bp1,
                           const float* __restrict__ Wp2, const float* __restrict__ bp2,
                           float* __restrict__ out) {
    int g = blockIdx.x, j = threadIdx.x;   // 64 threads
    const float* gr = gf + (size_t)g * (2 * HID);
    float acc = bp1[j];
    for (int k = 0; k < 2 * HID; k++) acc += gr[k] * Wp1[k * 64 + j];
    float y = acc * Wp2[j];
    __shared__ float red[64];
    red[j] = y;
    __syncthreads();
    for (int sft = 32; sft > 0; sft >>= 1) {
        if (j < sft) red[j] += red[j + sft];
        __syncthreads();
    }
    if (j == 0) out[g] = red[0] + bp2[0];
}

// ---------------- launcher ----------------

static inline size_t alup(size_t x) { return (x + 255) & ~(size_t)255; }

extern "C" void kernel_launch(void* const* d_in, const int* in_sizes, int n_in,
                              void* d_out, int out_size, void* d_ws, size_t ws_size,
                              hipStream_t stream) {
    const float* X       = (const float*)d_in[0];
    const int*   src     = (const int*)d_in[1];
    const int*   dst     = (const int*)d_in[2];
    const int*   gid     = (const int*)d_in[3];
    // d_in[4] = n_graphs (1024, compile-time constant here)
    const float* Wself1  = (const float*)d_in[5];
    const float* Wneigh1 = (const float*)d_in[6];
    const float* b1      = (const float*)d_in[7];
    const float* Wself2  = (const float*)d_in[8];
    const float* Wneigh2 = (const float*)d_in[9];
    const float* b2      = (const float*)d_in[10];
    const float* watom   = (const float*)d_in[11];
    const float* batom   = (const float*)d_in[12];
    const float* Wp1     = (const float*)d_in[13];
    const float* bp1     = (const float*)d_in[14];
    const float* Wp2     = (const float*)d_in[15];
    const float* bp2     = (const float*)d_in[16];

    char* w = (char*)d_ws;
    size_t o = 0;
    _Float16* X16  = (_Float16*)(w + o); o = alup(o + (size_t)NN * IN_P * 2);
    float*    rdeg = (float*)(w + o);    o = alup(o + (size_t)NN * 4);
    float*    agg  = (float*)(w + o);    o = alup(o + (size_t)NN * HID_P * 4);
    _Float16* h1   = (_Float16*)(w + o); o = alup(o + (size_t)NN * HID_P * 2);
    float*    h2   = (float*)(w + o);    o = alup(o + (size_t)NN * HID_P * 4);
    float*    gf   = (float*)(w + o);    o = alup(o + (size_t)NG * 2 * HID * 4);
    _Float16* Wt1  = (_Float16*)(w + o); o = alup(o + (size_t)HID_P * KTOT1 * 2);
    _Float16* Wt2  = (_Float16*)(w + o); o = alup(o + (size_t)HID_P * KTOT2 * 2);

    // zero accumulators used this call
    hipMemsetAsync(rdeg, 0, (size_t)NN * 4, stream);
    hipMemsetAsync(agg,  0, (size_t)NN * HID_P * 4, stream);
    hipMemsetAsync(gf,   0, (size_t)NG * 2 * HID * 4, stream);

    pad_x_kernel<<<(NN * IN_P + 255) / 256, 256, 0, stream>>>(X, X16);
    prep_w_kernel<<<(HID_P * KTOT2 + 255) / 256, 256, 0, stream>>>(
        Wself1, Wneigh1, Wself2, Wneigh2, Wt1, Wt2);
    deg_count_kernel<<<(NE + 255) / 256, 256, 0, stream>>>(dst, rdeg);
    rdeg_kernel<<<(NN + 255) / 256, 256, 0, stream>>>(rdeg);

    // layer 1: aggregate X, then fused GEMM -> h1 (f16)
    {
        int nc4 = IN_P / 4 - 1;                 // 19 chunks cover cols 0..75 (74,75 zero pad)
        int total = NE * nc4;
        scatter_kernel<<<(total + 255) / 256, 256, 0, stream>>>(
            X16, IN_P, nc4, src, dst, agg, HID_P, total);
    }
    sage_gemm_kernel<_Float16><<<(MT + 7) / 8, 256, 0, stream>>>(
        X16, IN_P, IN_P, agg, HID_P, rdeg, Wt1, KTOT1, b1, h1, HID);

    // layer 2: aggregate h1, fused GEMM -> h2 (f32)
    hipMemsetAsync(agg, 0, (size_t)NN * HID_P * 4, stream);
    {
        int nc4 = HID / 4;                      // 25 chunks = cols 0..99
        int total = NE * nc4;
        scatter_kernel<<<(total + 255) / 256, 256, 0, stream>>>(
            h1, HID_P, nc4, src, dst, agg, HID_P, total);
    }
    sage_gemm_kernel<float><<<(MT + 7) / 8, 256, 0, stream>>>(
        h1, HID_P, HID_P, agg, HID_P, rdeg, Wt2, KTOT2, b2, h2, HID);

    // readout + predictor
    readout_kernel<<<NN / 8, 256, 0, stream>>>(h2, gid, watom, batom, gf);
    mlp_kernel<<<NG, 64, 0, stream>>>(gf, Wp1, bp1, Wp2, bp2, (float*)d_out);
    (void)in_sizes; (void)n_in; (void)out_size; (void)ws_size;
}
